// TurboGeneQuantizer_30288109371661
// MI455X (gfx1250) — matrix operand order repaired and checked by reading, not compiled, not run on hardware
//
#include <hip/hip_runtime.h>
#include <cstdint>

// ---------------------------------------------------------------------------
// QJL quantizer pipeline for MI455X (gfx1250, wave32, WMMA).
//
// Per 128-token tile (one workgroup, 8 wave32s):
//   norms -> normalize(f16) -> GEMM1 (Xn @ R^T)  -> quantize/residual ->
//   GEMM2 (res @ JL^T) -> sign -> GEMM3 (sgn @ JL) -> combine ->
//   GEMM4 (approx @ R) -> rescale -> global store.
// All GEMMs: v_wmma_f32_16x16x32_f16, f32 accumulate, K=128 in 4 steps.
// x-tile is staged with GLOBAL_LOAD_ASYNC_TO_LDS_B128 (ASYNCcnt path);
// operand matrices staged in LDS in the layout that makes every WMMA
// B-fragment load two contiguous ds_load_b128 per lane.
// ---------------------------------------------------------------------------

typedef _Float16 half_t;
typedef __attribute__((ext_vector_type(16))) _Float16 v16h;
typedef __attribute__((ext_vector_type(8)))  float    v8f;

#define D_DIM   128
#define S_DIM   4096
#define H_DIM   32
#define B_DIM   2
#define C_DIM   8
#define TILE    128
#define QJL_K   1.2533141373155003f   // sqrt(pi/2)

// Async direct global->LDS copy, 16 bytes per lane (no VGPR data round-trip).
// GV mode: vdst = LDS byte address, vaddr = 64-bit global address, saddr=off.
// Tracked by ASYNCcnt; pair with s_wait_asynccnt + workgroup barrier.
__device__ __forceinline__ void async_load_lds_b128(unsigned lds_byte_addr,
                                                    const void* gaddr) {
  asm volatile("global_load_async_to_lds_b128 %0, %1, off"
               :: "v"(lds_byte_addr), "v"(gaddr)
               : "memory");
}

__device__ __forceinline__ void wait_asynccnt0() {
  asm volatile("s_wait_asynccnt 0x0" ::: "memory");
}

// A fragment: 16x32 f16 (wave32 layout, CDNA5 ISA 7.12.2).
// Lane L: M = L&15, hs = L>>4. V0-3 hold K = k0+8hs .. +7 (pairs),
// V4-7 hold K = k0+16+8hs .. +7.  buf is row-major [row][k], stride 128.
__device__ __forceinline__ v16h load_frag_a(const half_t* __restrict__ buf,
                                            int row0, int k0, int lane) {
  const int M  = lane & 15;
  const int hs = lane >> 4;
  const half_t* p = buf + (size_t)(row0 + M) * D_DIM + k0 + 8 * hs;
  union { v16h v; uint4 q[2]; } r;
  r.q[0] = *(const uint4*)(p);
  r.q[1] = *(const uint4*)(p + 16);
  return r.v;
}

// B fragment: 32x16 f16. Lane L: N = L&15, hs = L>>4; holds K = k0+8hs .. +7
// and K = k0+16+8hs .. +7 for its column (same as A).  Storage S must satisfy
// S[n*128 + k] == B[k][n]; then the load is two 16-byte runs, 32 bytes apart.
__device__ __forceinline__ v16h load_frag_b(const half_t* __restrict__ buf,
                                            int n0, int k0, int lane) {
  const int N  = lane & 15;
  const int hs = lane >> 4;
  const half_t* p = buf + (size_t)(n0 + N) * D_DIM + k0 + 8 * hs;
  union { v16h v; uint4 q[2]; } r;
  r.q[0] = *(const uint4*)(p);
  r.q[1] = *(const uint4*)(p + 16);
  return r.v;
}

// 128x128x128 GEMM; wave w owns row-tile w, iterates 8 column tiles;
// K unrolled into 4 wmma ops per tile.  Epilogue gets (tokenRow, col, value)
// with D layout M = r + 8*(lane>>4), N = lane&15 (ISA 7.12.2).
template <typename Epi>
__device__ __forceinline__ void gemm128(const half_t* __restrict__ A,
                                        const half_t* __restrict__ Bm,
                                        int wave, int lane, Epi epi) {
  const int row0 = wave * 16;
  v16h afr[4];
#pragma unroll
  for (int kk = 0; kk < 4; ++kk) afr[kk] = load_frag_a(A, row0, kk * 32, lane);
  const int hs = lane >> 4;
  const int Nl = lane & 15;
#pragma unroll
  for (int ct = 0; ct < 8; ++ct) {
    v8f acc = {};
#pragma unroll
    for (int kk = 0; kk < 4; ++kk) {
      v16h bfr = load_frag_b(Bm, ct * 16, kk * 32, lane);
      acc = __builtin_amdgcn_wmma_f32_16x16x32_f16(
          /*neg_a=*/false, afr[kk], /*neg_b=*/false, bfr,
          /*c_mod=*/(short)0, acc, /*reuse_a=*/false, /*reuse_b=*/false);
    }
#pragma unroll
    for (int r = 0; r < 8; ++r) {
      epi(row0 + r + 8 * hs, ct * 16 + Nl, acc[r]);
    }
  }
}

__device__ __forceinline__ float nearest_centroid(float v, const float c[8]) {
  float best = c[0];
  float bd = fabsf(v - c[0]);
#pragma unroll
  for (int j = 1; j < 8; ++j) {
    float d = fabsf(v - c[j]);
    if (d < bd) { bd = d; best = c[j]; }   // strict '<' keeps first on ties
  }
  return best;
}

__global__ __launch_bounds__(256)
void qjl_pipeline_kernel(const float* __restrict__ x,
                         const float* __restrict__ rot,
                         const float* __restrict__ jl,
                         const float* __restrict__ cent,
                         float* __restrict__ out) {
  // --- LDS (~259 KB of 320 KB/WGP) ---
  __shared__ __align__(16) half_t sRdir[D_DIM * D_DIM];  // [e][d] = R[e][d]
  __shared__ __align__(16) half_t sRtr [D_DIM * D_DIM];  // [i][j] = R[j][i]
  __shared__ __align__(16) half_t sJdir[D_DIM * D_DIM];  // [e][d] = JL[e][d]
  __shared__ __align__(16) half_t sJtr [D_DIM * D_DIM];  // [i][j] = JL[j][i]
  __shared__ __align__(16) half_t sA16 [TILE * D_DIM];   // GEMM A staging
  __shared__ __align__(16) half_t sDq16[TILE * D_DIM];   // dequantized unit vecs
  __shared__ __align__(16) float  sXf  [TILE * D_DIM];   // f32 work tile (64 KB)
  __shared__ float sPart[256];
  __shared__ float sNorm[TILE];
  __shared__ float sInv [TILE];
  __shared__ float sScale[TILE];
  __shared__ float sCent[C_DIM];

  const int tid  = threadIdx.x;
  const int wave = tid >> 5;
  const int lane = tid & 31;

  const int tilesPerBH = S_DIM / TILE;            // 32
  const int bh = blockIdx.x / tilesPerBH;         // b*H + h
  const int h  = bh % H_DIM;
  const int s0 = (blockIdx.x % tilesPerBH) * TILE;

  const float* __restrict__ xTile   = x   + ((size_t)bh * S_DIM + s0) * D_DIM;
  float*       __restrict__ outTile = out + ((size_t)bh * S_DIM + s0) * D_DIM;
  const float* __restrict__ Rh      = rot + (size_t)h * D_DIM * D_DIM;

  // Kick the 64 KB x-tile into LDS with the async DMA path (no VGPR staging).
  // 4096 b128 transfers, 16 per thread, fully coalesced.
  {
    const unsigned ldsX = (unsigned)(uintptr_t)sXf;
    const char* g = (const char*)xTile;
    for (int i = tid; i < TILE * D_DIM / 4; i += 256) {
      async_load_lds_b128(ldsX + (unsigned)i * 16u, g + (size_t)i * 16u);
    }
  }

  // Prefetch R_h / JL cachelines (global_prefetch_b8) ahead of staging.
  __builtin_prefetch((const char*)Rh + (size_t)tid * 64, 0, 0);
  __builtin_prefetch((const char*)jl + (size_t)tid * 64, 0, 0);

  if (tid < C_DIM) sCent[tid] = cent[h * C_DIM + tid];

  // Stage R_h and JL as f16 in both direct and transposed layouts so every
  // WMMA B-fragment load below is contiguous.  float4 loads, packed direct
  // stores; transposed stores are scattered but run once per block.
  for (int i = tid; i < (D_DIM * D_DIM) / 4; i += 256) {
    const float4 r4 = ((const float4*)Rh)[i];
    const float4 j4 = ((const float4*)jl)[i];
    const int base = i * 4;
    const int e = base >> 7;
    const int d = base & 127;
    const half_t rh[4] = {(half_t)r4.x, (half_t)r4.y, (half_t)r4.z, (half_t)r4.w};
    const half_t jh[4] = {(half_t)j4.x, (half_t)j4.y, (half_t)j4.z, (half_t)j4.w};
#pragma unroll
    for (int q = 0; q < 4; ++q) {
      sRdir[base + q] = rh[q];            // S1[n=e][k=d] = R[e][d]   (GEMM1)
      sJdir[base + q] = jh[q];            // S2[n=e][k=d] = JL[e][d]  (GEMM2)
      sRtr[(d + q) * D_DIM + e] = rh[q];  // S4[n=e][k=d] = R[d][e]   (GEMM4)
      sJtr[(d + q) * D_DIM + e] = jh[q];  // S3[n=d][k=e] = JL[e][d]  (GEMM3)
    }
  }

  wait_asynccnt0();   // this wave's async x-tile transfers complete
  __syncthreads();    // all waves' transfers + matrix staging visible

  float cr[C_DIM];
#pragma unroll
  for (int j = 0; j < C_DIM; ++j) cr[j] = sCent[j];

  const int t  = tid & 127;   // token within tile (2 threads per token)
  const int hh = tid >> 7;    // which half of the 128 dims

  // ---- vector norms + normalize into f16 A ----
  {
    const float* row = sXf + t * D_DIM + hh * 64;
    float s = 0.f;
#pragma unroll 8
    for (int k = 0; k < 64; ++k) { const float v = row[k]; s += v * v; }
    sPart[tid] = s;
  }
  __syncthreads();
  if (tid < TILE) {
    const float nv = sqrtf(sPart[tid] + sPart[tid + 128]);
    sNorm[tid] = nv;
    sInv[tid]  = (nv > 0.f) ? (1.f / nv) : 1.f;
  }
  __syncthreads();
  {
    const float inv = sInv[t];
    const float* row = sXf + t * D_DIM + hh * 64;
    half_t* dr = sA16 + t * D_DIM + hh * 64;
#pragma unroll 8
    for (int k = 0; k < 64; ++k) dr[k] = (half_t)(row[k] * inv);
  }
  __syncthreads();

  // ---- GEMM1: x_rot = Xn @ R^T  -> sXf ----
  gemm128(sA16, sRdir, wave, lane,
          [&](int tr, int c, float v) { sXf[tr * D_DIM + c] = v; });
  __syncthreads();

  // ---- quantize to nearest centroid, deq norm, residual ----
  {
    const float* row = sXf + t * D_DIM + hh * 64;
    float s = 0.f;
#pragma unroll 4
    for (int k = 0; k < 64; ++k) {
      const float q = nearest_centroid(row[k], cr);
      s += q * q;
    }
    sPart[tid] = s;
  }
  __syncthreads();
  if (tid < TILE) {
    const float dn = sqrtf(sPart[tid] + sPart[tid + 128]);
    sInv[tid] = (dn > 1e-10f) ? (1.f / dn) : 1.f;
  }
  __syncthreads();
  {
    const float invd = sInv[t];
    const float* row = sXf + t * D_DIM + hh * 64;
    half_t* dqr = sDq16 + t * D_DIM + hh * 64;
    half_t* rr  = sA16  + t * D_DIM + hh * 64;
    float s = 0.f;
#pragma unroll 4
    for (int k = 0; k < 64; ++k) {
      const float v = row[k];
      const float u = nearest_centroid(v, cr) * invd;  // x_deq_unit
      const float res = v - u;                         // residual
      s += res * res;
      dqr[k] = (half_t)u;
      rr[k]  = (half_t)res;
    }
    sPart[tid] = s;
  }
  __syncthreads();
  if (tid < TILE) {
    const float rn = sqrtf(sPart[tid] + sPart[tid + 128]);
    sScale[tid] = (QJL_K / (float)D_DIM) * rn;
  }
  __syncthreads();

  // ---- GEMM2: projected = residual @ JL^T -> signs (reuse sXf as f16) ----
  half_t* sSgn = (half_t*)sXf;   // x_rot no longer needed
  gemm128(sA16, sJdir, wave, lane, [&](int tr, int c, float v) {
    sSgn[tr * D_DIM + c] = (half_t)((v > 0.f) ? 1.f : -1.f);
  });
  __syncthreads();

  // ---- GEMM3: correction = sgn @ JL; approx = dq_unit + corr*scale -> sA16 ----
  gemm128(sSgn, sJtr, wave, lane, [&](int tr, int c, float v) {
    const float ap = (float)sDq16[tr * D_DIM + c] + v * sScale[tr];
    sA16[tr * D_DIM + c] = (half_t)ap;
  });
  __syncthreads();

  // ---- GEMM4: recon = approx @ R; rescale by norm; store to global ----
  gemm128(sA16, sRtr, wave, lane, [&](int tr, int c, float v) {
    outTile[(size_t)tr * D_DIM + c] = v * sNorm[tr];
  });
}

extern "C" void kernel_launch(void* const* d_in, const int* in_sizes, int n_in,
                              void* d_out, int out_size, void* d_ws, size_t ws_size,
                              hipStream_t stream) {
  (void)in_sizes; (void)n_in; (void)out_size; (void)d_ws; (void)ws_size;
  const float* x    = (const float*)d_in[0];
  const float* rot  = (const float*)d_in[1];
  const float* jlm  = (const float*)d_in[2];
  const float* cent = (const float*)d_in[3];
  float* out = (float*)d_out;

  dim3 grid(B_DIM * H_DIM * (S_DIM / TILE));  // 2048 workgroups
  dim3 block(256);                            // 8 wave32s
  hipLaunchKernelGGL(qjl_pipeline_kernel, grid, block, 0, stream,
                     x, rot, jlm, cent, out);
}
